// Cross_VQEmbeddingEMA_AV_hierarchical_1_43044162240534
// MI455X (gfx1250) — matrix operand order repaired and checked by reading, not compiled, not run on hardware
//
#include <hip/hip_runtime.h>

#define B_  64
#define T_  512
#define D_  512
#define M_  1024
#define BT_ (B_*T_)          // 32768
#define DT_ (2*D_)           // 1024 (full embedding width)

typedef __attribute__((ext_vector_type(16))) __bf16 v16bf;
typedef __attribute__((ext_vector_type(8)))  __bf16 v8bf;
typedef __attribute__((ext_vector_type(8)))  float  v8f;
typedef __attribute__((ext_vector_type(4)))  unsigned short us4;

__device__ __forceinline__ unsigned short f2bf(float f) {
  unsigned int u = __float_as_uint(f);
  u += 0x7FFFu + ((u >> 16) & 1u);       // round-to-nearest-even
  return (unsigned short)(u >> 16);
}

// ---------------------------------------------------------------- k_zero ----
__global__ void k_zero(float* __restrict__ p, int n) {
  int i = blockIdx.x * 256 + threadIdx.x;
  if (i < n) p[i] = 0.f;
}

// --------------------------------------------------------------- k_embed ----
// Split embedding into bf16 video/audio codebooks + squared norms.
__global__ void k_embed(const float* __restrict__ emb,
                        unsigned short* __restrict__ eV,
                        unsigned short* __restrict__ eA,
                        float* __restrict__ eNv, float* __restrict__ eNa) {
  int m = blockIdx.x, tid = threadIdx.x;
  float nv = 0.f, na = 0.f;
  for (int k = tid; k < D_; k += 256) {
    float v = emb[(size_t)m * DT_ + k];
    float a = emb[(size_t)m * DT_ + D_ + k];
    eV[(size_t)m * D_ + k] = f2bf(v);
    eA[(size_t)m * D_ + k] = f2bf(a);
    nv += v * v; na += a * a;
  }
  __shared__ float sv[256], sa[256];
  sv[tid] = nv; sa[tid] = na; __syncthreads();
  for (int s = 128; s > 0; s >>= 1) {
    if (tid < s) { sv[tid] += sv[tid + s]; sa[tid] += sa[tid + s]; }
    __syncthreads();
  }
  if (tid == 0) { eNv[m] = sv[0]; eNa[m] = sa[0]; }
}

// ---------------------------------------------------------------- k_dist ----
// Fused: bf16 WMMA x.e^T  ->  distances -> argmin -> softmax stats -> pH
//        atomics -> histogram -> gather outputs.  16 rows per block, 8 waves,
//        each wave owns a 16x128 column slice (8 WMMA accumulators).
__global__ void __launch_bounds__(256)
k_dist(const float* __restrict__ x, const float* __restrict__ emb,
       const unsigned short* __restrict__ eB, const float* __restrict__ eNorm,
       float* __restrict__ pH, int* __restrict__ hist,
       float* __restrict__ out_full, float* __restrict__ out_quant,
       int halfSel) {
  constexpr int LDA = D_ + 8;                 // ushort stride, pad vs banks
  __shared__ unsigned short sA[16 * LDA];     // 16x512 bf16 A panel
  __shared__ float sXn[16];
  __shared__ float sWMin[8][16];  __shared__ int sWArg[8][16];
  __shared__ float sWSum[8][16];
  __shared__ float sRMin[16];     __shared__ int sRArg[16];
  __shared__ float sRSum[16];

  const int tid  = threadIdx.x;
  const int row0 = blockIdx.x * 16;           // 16 rows, all in same batch b
  const int b    = row0 / T_;

  // ---- phase 0: stage A panel (f32 -> bf16, b128 traffic) + row norms -----
  {
    int g = tid >> 4, l = tid & 15;
    const float4* xr = (const float4*)(x + (size_t)(row0 + g) * D_);
    float nrm = 0.f;
    #pragma unroll
    for (int j = 0; j < D_ / 64; j++) {       // 8 iters * 16 lanes * 4 = 512
      float4 v = xr[l + 16 * j];
      nrm += v.x * v.x + v.y * v.y + v.z * v.z + v.w * v.w;
      us4 h;
      h.x = f2bf(v.x); h.y = f2bf(v.y); h.z = f2bf(v.z); h.w = f2bf(v.w);
      *(us4*)(&sA[g * LDA + (l + 16 * j) * 4]) = h;
    }
    #pragma unroll
    for (int mk = 8; mk; mk >>= 1) nrm += __shfl_xor(nrm, mk, 16);
    if (l == 0) sXn[g] = nrm;
  }
  __syncthreads();

  const int wave = tid >> 5, lane = tid & 31;
  const int hl = lane >> 4, ln = lane & 15;
  const int nwave = wave * 128;

  float en[8];
  #pragma unroll
  for (int t = 0; t < 8; t++) en[t] = eNorm[nwave + t * 16 + ln];

  v8f acc[8];
  #pragma unroll
  for (int t = 0; t < 8; t++) {
    v8f z = {0.f, 0.f, 0.f, 0.f, 0.f, 0.f, 0.f, 0.f};
    acc[t] = z;
  }

  // ---- phase 1: GEMM (16x512)x(512x128 slice), 16 k-steps of k=32 ---------
  // All 8 B fragments for a k-step are preloaded into distinct registers so
  // the 16 global_load_b128 issue as one clause and overlap the WMMA chain.
  for (int k0 = 0; k0 < D_; k0 += 32) {
    // A fragment per ISA 16-bit 16x32 layout:
    //   lanes 0-15 : row=lane, K = {k0..k0+7, k0+16..k0+23}
    //   lanes 16-31: K = {k0+8..15, k0+24..31}
    union { v16bf v; v8bf h[2]; } af;
    af.h[0] = *(const v8bf*)(&sA[ln * LDA + k0 + hl * 8]);
    af.h[1] = *(const v8bf*)(&sA[ln * LDA + k0 + 16 + hl * 8]);

    union { v16bf v; v8bf h[2]; } bfr[8];
    #pragma unroll
    for (int t = 0; t < 8; t++) {
      // B fragment per ISA 16-bit 32x16 layout: col = lane&15,
      // K = {k0+(lane>>4)*8 .. +7, k0+16+(lane>>4)*8 .. +7}
      const unsigned short* bp =
          eB + (size_t)(nwave + t * 16 + ln) * D_ + k0 + hl * 8;
      bfr[t].h[0] = *(const v8bf*)(bp);
      bfr[t].h[1] = *(const v8bf*)(bp + 16);
    }
    if (k0 + 32 < D_)
      __builtin_prefetch(eB + (size_t)(nwave + ln) * D_ + k0 + 32, 0, 1);
    #pragma unroll
    for (int t = 0; t < 8; t++)
      acc[t] = __builtin_amdgcn_wmma_f32_16x16x32_bf16(
          false, af.v, false, bfr[t].v, (short)0, acc[t], false, false);
  }

  // ---- phase 2: distances + per-row min / argmin --------------------------
  // C/D layout: lane holds col n = base+(lane&15); VGPR r -> row r+8*(lane>>4)
  float xn[8];
  #pragma unroll
  for (int r = 0; r < 8; r++) xn[r] = sXn[r + 8 * hl];

  float rmin[8]; int rarg[8];
  #pragma unroll
  for (int r = 0; r < 8; r++) { rmin[r] = 3.4e38f; rarg[r] = 0; }

  #pragma unroll
  for (int t = 0; t < 8; t++) {
    int n = nwave + t * 16 + ln;
    #pragma unroll
    for (int r = 0; r < 8; r++) {
      float d = xn[r] + en[t] - 2.f * acc[t][r];
      acc[t][r] = d;                         // keep distances in registers
      if (d < rmin[r]) { rmin[r] = d; rarg[r] = n; }
    }
  }
  #pragma unroll
  for (int r = 0; r < 8; r++) {
    #pragma unroll
    for (int mk = 1; mk < 16; mk <<= 1) {
      float od = __shfl_xor(rmin[r], mk, 16);
      int   oa = __shfl_xor(rarg[r], mk, 16);
      if (od < rmin[r] || (od == rmin[r] && oa < rarg[r])) {
        rmin[r] = od; rarg[r] = oa;
      }
    }
    if (ln == 0) { sWMin[wave][r + 8 * hl] = rmin[r];
                   sWArg[wave][r + 8 * hl] = rarg[r]; }
  }
  __syncthreads();
  if (tid < 16) {
    float bm = 3.4e38f; int ba = 0;
    for (int w = 0; w < 8; w++) {
      float dw = sWMin[w][tid]; int aw = sWArg[w][tid];
      if (dw < bm || (dw == bm && aw < ba)) { bm = dw; ba = aw; }
    }
    sRMin[tid] = bm; sRArg[tid] = ba;
  }
  __syncthreads();

  // ---- phase 3: stable softmax row sums: p = exp(sqrt(dmin)-sqrt(d)) ------
  float sm[8], psum[8];
  #pragma unroll
  for (int r = 0; r < 8; r++) {
    sm[r] = sqrtf(fmaxf(sRMin[r + 8 * hl], 0.f));
    psum[r] = 0.f;
  }
  #pragma unroll
  for (int t = 0; t < 8; t++)
    #pragma unroll
    for (int r = 0; r < 8; r++) {
      float pv = __expf(sm[r] - sqrtf(fmaxf(acc[t][r], 0.f)));
      acc[t][r] = pv;
      psum[r] += pv;
    }
  #pragma unroll
  for (int r = 0; r < 8; r++) {
    #pragma unroll
    for (int mk = 8; mk; mk >>= 1) psum[r] += __shfl_xor(psum[r], mk, 16);
    if (ln == 0) sWSum[wave][r + 8 * hl] = psum[r];
  }
  __syncthreads();
  if (tid < 16) {
    float s = 0.f;
    for (int w = 0; w < 8; w++) s += sWSum[w][tid];
    sRSum[tid] = s;
  }
  __syncthreads();

  // ---- phase 4: column sums of normalized p -> pH accumulation ------------
  float inv[8];
  #pragma unroll
  for (int r = 0; r < 8; r++) inv[r] = 1.f / sRSum[r + 8 * hl];
  #pragma unroll
  for (int t = 0; t < 8; t++) {
    float c = 0.f;
    #pragma unroll
    for (int r = 0; r < 8; r++) c += acc[t][r] * inv[r];
    atomicAdd(&pH[(size_t)b * M_ + nwave + t * 16 + ln], c);
  }

  // ---- phase 5: histogram + gathers (b128 traffic) ------------------------
  if (tid < 16) atomicAdd(&hist[(size_t)b * M_ + sRArg[tid]], 1);
  for (int m = 0; m < 16; m++) {
    int idx = sRArg[m];
    size_t grow = (size_t)(row0 + m);
    const float4* src4 = (const float4*)(emb + (size_t)idx * DT_);
    float4* df4 = (float4*)(out_full + grow * DT_);
    #pragma unroll
    for (int i = tid; i < DT_ / 4; i += 256) df4[i] = src4[i];   // 1 iter
    const float4* sq4 = (const float4*)(emb + (size_t)idx * DT_ + halfSel * D_);
    float4* dq4 = (float4*)(out_quant + grow * D_);
    if (tid < D_ / 4) dq4[tid] = sq4[tid];                        // 128 lanes
  }
}

// ---------------------------------------------------------------- k_logs ----
__global__ void k_logs(const float* __restrict__ vpH, const float* __restrict__ apH,
                       float* __restrict__ vN, float* __restrict__ aN,
                       float* __restrict__ logV, float* __restrict__ logA) {
  int i = blockIdx.x * 256 + threadIdx.x;   // B_*M_ total
  float vm = vpH[i] * (1.f / T_);
  float am = apH[i] * (1.f / T_);
  vN[i] = vm; aN[i] = am;
  logV[i] = __logf(vm + 1e-10f);
  logA[i] = __logf(am + 1e-10f);
}

// --------------------------------------------------------------- k_scode ----
__global__ void k_scode(const float* __restrict__ vN, const float* __restrict__ aN,
                        const float* __restrict__ logV, const float* __restrict__ logA,
                        float* __restrict__ Sc) {
  int i = blockIdx.x, j = threadIdx.x;      // 64 x 64
  float s = 0.f;
  for (int m = 0; m < M_; m++)
    s += aN[i * M_ + m] * logV[j * M_ + m] + vN[i * M_ + m] * logA[j * M_ + m];
  Sc[i * 64 + j] = s;
}

// --------------------------------------------------------------- k_final ----
__global__ void k_final(const float* __restrict__ Sc, float* __restrict__ out_l) {
  __shared__ float red[256];
  int tid = threadIdx.x;
  float mn = 3.4e38f;
  for (int i = tid; i < 64 * 64; i += 256) mn = fminf(mn, Sc[i]);
  red[tid] = mn; __syncthreads();
  for (int s = 128; s > 0; s >>= 1) {
    if (tid < s) red[tid] = fminf(red[tid], red[tid + s]);
    __syncthreads();
  }
  float Max = -red[0];                      // MaxScode = max(-S) = -min(S)
  __syncthreads();
  float v = 0.f;
  if (tid < 64) {
    float rs = 0.f;
    for (int j = 0; j < 64; j++) rs += __expf(Sc[tid * 64 + j] + Max);
    v = __logf(__expf(Sc[tid * 64 + tid] + Max) / (rs + 1e-5f));
  }
  red[tid] = v; __syncthreads();
  for (int s = 128; s > 0; s >>= 1) {
    if (tid < s) red[tid] += red[tid + s];
    __syncthreads();
  }
  if (tid == 0) out_l[0] = -red[0] / 64.f;
}

// ---------------------------------------------------------------- k_mode ----
__global__ void k_mode(const int* __restrict__ vh, const int* __restrict__ ah,
                       float* __restrict__ out_e) {
  __shared__ int bc[256], bi[256];
  int tid = threadIdx.x;
  int eq = 0;
  for (int b = 0; b < B_; b++) {
    int c0 = -1, i0 = 0;
    for (int m = tid; m < M_; m += 256) {
      int c = vh[b * M_ + m];
      if (c > c0 || (c == c0 && m < i0)) { c0 = c; i0 = m; }
    }
    bc[tid] = c0; bi[tid] = i0; __syncthreads();
    for (int s = 128; s > 0; s >>= 1) {
      if (tid < s && (bc[tid + s] > bc[tid] ||
          (bc[tid + s] == bc[tid] && bi[tid + s] < bi[tid]))) {
        bc[tid] = bc[tid + s]; bi[tid] = bi[tid + s];
      }
      __syncthreads();
    }
    int vm = bi[0]; __syncthreads();
    c0 = -1; i0 = 0;
    for (int m = tid; m < M_; m += 256) {
      int c = ah[b * M_ + m];
      if (c > c0 || (c == c0 && m < i0)) { c0 = c; i0 = m; }
    }
    bc[tid] = c0; bi[tid] = i0; __syncthreads();
    for (int s = 128; s > 0; s >>= 1) {
      if (tid < s && (bc[tid + s] > bc[tid] ||
          (bc[tid + s] == bc[tid] && bi[tid + s] < bi[tid]))) {
        bc[tid] = bc[tid + s]; bi[tid] = bi[tid + s];
      }
      __syncthreads();
    }
    int am = bi[0]; __syncthreads();
    if (vm == am) eq++;
  }
  if (tid == 0) out_e[0] = (float)eq;
}

// ----------------------------------------------------------------------------
extern "C" void kernel_launch(void* const* d_in, const int* in_sizes, int n_in,
                              void* d_out, int out_size, void* d_ws, size_t ws_size,
                              hipStream_t stream) {
  const float* audio = (const float*)d_in[0];
  const float* video = (const float*)d_in[1];
  const float* emb   = (const float*)d_in[2];
  // d_in[3] modality_weights, d_in[4] epoch: unused by the reference output.
  float* out = (float*)d_out;

  char* ws = (char*)d_ws;
  unsigned short* eV   = (unsigned short*)(ws);                 // 1,048,576 B
  unsigned short* eA   = (unsigned short*)(ws + 1048576);       // 1,048,576 B
  float* eNv   = (float*)(ws + 2097152);                        // 4 KB
  float* eNa   = (float*)(ws + 2101248);                        // 4 KB
  float* vpH   = (float*)(ws + 2105344);                        // 256 KB
  float* apH   = (float*)(ws + 2367488);                        // 256 KB
  int*   vhist = (int*)  (ws + 2629632);                        // 256 KB
  int*   ahist = (int*)  (ws + 2891776);                        // 256 KB
  float* vN    = (float*)(ws + 3153920);
  float* aN    = (float*)(ws + 3416064);
  float* logV  = (float*)(ws + 3678208);
  float* logA  = (float*)(ws + 3940352);
  float* Sc    = (float*)(ws + 4202496);                        // 16 KB

  float* out_vfull = out;
  float* out_afull = out + (size_t)BT_ * DT_;
  float* out_vq    = out + (size_t)BT_ * DT_ * 2;
  float* out_aq    = out_vq + (size_t)BT_ * D_;
  float* out_l     = out_aq + (size_t)BT_ * D_;
  float* out_e     = out_l + 1;

  // Zero pH sums + histograms (contiguous 4*B*M words).
  k_zero<<<dim3((4 * B_ * M_ + 255) / 256), dim3(256), 0, stream>>>(vpH, 4 * B_ * M_);
  k_embed<<<dim3(M_), dim3(256), 0, stream>>>(emb, eV, eA, eNv, eNa);
  k_dist<<<dim3(BT_ / 16), dim3(256), 0, stream>>>(
      video, emb, eV, eNv, vpH, vhist, out_vfull, out_vq, 0);
  k_dist<<<dim3(BT_ / 16), dim3(256), 0, stream>>>(
      audio, emb, eA, eNa, apH, ahist, out_afull, out_aq, 1);
  k_logs<<<dim3(B_ * M_ / 256), dim3(256), 0, stream>>>(vpH, apH, vN, aN, logV, logA);
  k_scode<<<dim3(64), dim3(64), 0, stream>>>(vN, aN, logV, logA, Sc);
  k_final<<<dim3(1), dim3(256), 0, stream>>>(Sc, out_l);
  k_mode<<<dim3(1), dim3(256), 0, stream>>>(vhist, ahist, out_e);
}
